// EGNNAreaPlusHuman_45578192945210
// MI455X (gfx1250) — matrix operand order repaired and checked by reading, not compiled, not run on hardware
//
#include <hip/hip_runtime.h>
#include <hip/hip_bf16.h>
#include <float.h>

// ---------------------------------------------------------------------------
// EGNN mesh network for MI455X (gfx1250, wave32, WMMA).
// All dense layers run through v_wmma_f32_16x16x32_bf16 (fp32 -> bf16 RNE,
// f32 accumulate). Edge gathers are fused into the GEMM A-tile loads.
// Weights are stored bf16 *transposed* ([Nc][K]); the B tile is brought into
// LDS with GLOBAL_LOAD_ASYNC_TO_LDS_B128 (ASYNCcnt path), A tile goes through
// VGPRs (needs f32->bf16 conversion). Segment sums/max use f32 global atomics.
// conv3 coordinate branch is dead w.r.t. the returned log-softmax -> skipped.
// ---------------------------------------------------------------------------

#define NN 20000
#define NE 120000
#define NF 40000

typedef __attribute__((ext_vector_type(16))) __bf16 v16bf;
typedef __attribute__((ext_vector_type(8)))  __bf16 v8bf;
typedef __attribute__((ext_vector_type(8)))  float  v8f;

static __device__ __forceinline__ unsigned short f32_to_bf16_bits(float f) {
  unsigned int u = __float_as_uint(f);
  u += 0x7FFFu + ((u >> 16) & 1u);          // round-to-nearest-even
  return (unsigned short)(u >> 16);
}
static __device__ __forceinline__ uint2 pack4_bf16(float4 v) {
  uint2 r;
  r.x = (unsigned)f32_to_bf16_bits(v.x) | ((unsigned)f32_to_bf16_bits(v.y) << 16);
  r.y = (unsigned)f32_to_bf16_bits(v.z) | ((unsigned)f32_to_bf16_bits(v.w) << 16);
  return r;
}
static __device__ __forceinline__ float silu(float v) {
  return v / (1.0f + __expf(-v));
}
static __device__ __forceinline__ void atomicMaxF32(float* addr, float val) {
  if (val >= 0.0f) atomicMax((int*)addr, __float_as_int(val));
  else             atomicMin((unsigned int*)addr, __float_as_uint(val));
}

// ------------------------------ tiny utility kernels -----------------------
// fp32 [K,Nc] row-major -> bf16 [Nc,K] (K-contiguous) for b128 B-tile loads
__global__ void k_f32_to_bf16_T(const float* __restrict__ src,
                                unsigned short* __restrict__ dst, int K, int Nc) {
  int i = blockIdx.x * blockDim.x + threadIdx.x;
  if (i >= K * Nc) return;
  int k = i / Nc, n = i - k * Nc;
  dst[(size_t)n * K + k] = f32_to_bf16_bits(src[i]);
}
__global__ void k_zero_f32(float* __restrict__ p, int n) {
  int i = blockIdx.x * blockDim.x + threadIdx.x;
  if (i < n) p[i] = 0.0f;
}
__global__ void k_zero_i32(int* __restrict__ p, int n) {
  int i = blockIdx.x * blockDim.x + threadIdx.x;
  if (i < n) p[i] = 0;
}
__global__ void k_fill_bits(unsigned int* __restrict__ p, unsigned int v, int n) {
  int i = blockIdx.x * blockDim.x + threadIdx.x;
  if (i < n) p[i] = v;
}
__global__ void k_copy_f32(const float* __restrict__ s, float* __restrict__ d, int n) {
  int i = blockIdx.x * blockDim.x + threadIdx.x;
  if (i < n) d[i] = s[i];
}

// x0[n,c] = b[c] + area[n]*W[0,c] + sum_k hks[n,k]*W[1+k,c]   (W is 10x64)
__global__ void k_feat(const float* __restrict__ area, const float* __restrict__ hks,
                       const float* __restrict__ W, const float* __restrict__ b,
                       float* __restrict__ out) {
  int i = blockIdx.x * blockDim.x + threadIdx.x;
  if (i >= NN * 64) return;
  int n = i >> 6, c = i & 63;
  float v = b[c] + area[n] * W[c];
  #pragma unroll
  for (int k = 0; k < 9; ++k) v += hks[n * 9 + k] * W[(k + 1) * 64 + c];
  out[i] = v;
}

__global__ void k_cnt(const int* __restrict__ row, int* __restrict__ cnt) {
  int e = blockIdx.x * blockDim.x + threadIdx.x;
  if (e < NE) atomicAdd(&cnt[row[e]], 1);
}

__global__ void k_radial(const float* __restrict__ coord, const int* __restrict__ row,
                         const int* __restrict__ col, float* __restrict__ radial) {
  int e = blockIdx.x * blockDim.x + threadIdx.x;
  if (e >= NE) return;
  int r = row[e], c = col[e];
  float dx = coord[r * 3 + 0] - coord[c * 3 + 0];
  float dy = coord[r * 3 + 1] - coord[c * 3 + 1];
  float dz = coord[r * 3 + 2] - coord[c * 3 + 2];
  radial[e] = dx * dx + dy * dy + dz * dz;
}

// esc[e] = dot(cm[e,:hid], c2W[:hid])   (c2 has no bias)
__global__ void k_esc(const float* __restrict__ cm, const float* __restrict__ c2W,
                      int hid, float* __restrict__ esc) {
  int e = blockIdx.x * blockDim.x + threadIdx.x;
  if (e >= NE) return;
  const float* rowp = cm + (size_t)e * hid;
  float s = 0.0f;
  for (int k = 0; k < hid; ++k) s += rowp[k] * c2W[k];
  esc[e] = s;
}

__global__ void k_coord_scatter(const float* __restrict__ coord, const int* __restrict__ row,
                                const int* __restrict__ col, const float* __restrict__ esc,
                                float* __restrict__ acc) {
  int e = blockIdx.x * blockDim.x + threadIdx.x;
  if (e >= NE) return;
  int r = row[e], c = col[e];
  float s = esc[e];
  atomicAdd(&acc[r * 3 + 0], (coord[r * 3 + 0] - coord[c * 3 + 0]) * s);
  atomicAdd(&acc[r * 3 + 1], (coord[r * 3 + 1] - coord[c * 3 + 1]) * s);
  atomicAdd(&acc[r * 3 + 2], (coord[r * 3 + 2] - coord[c * 3 + 2]) * s);
}

__global__ void k_coord_update(const float* __restrict__ cin, const float* __restrict__ acc,
                               const int* __restrict__ cnt, float* __restrict__ cout) {
  int i = blockIdx.x * blockDim.x + threadIdx.x;
  if (i >= NN * 3) return;
  float c = fmaxf((float)cnt[i / 3], 1.0f);
  cout[i] = cin[i] + acc[i] / c;
}

__global__ void k_agg_scatter(const float* __restrict__ m, const int* __restrict__ row,
                              int hid, float* __restrict__ agg, int total) {
  int i = blockIdx.x * blockDim.x + threadIdx.x;
  if (i >= total) return;
  int e = i / hid, c = i - e * hid;
  atomicAdd(&agg[(size_t)row[e] * hid + c], m[i]);
}

// face max pool: 3F (vertex,face) pairs x 512 channels
__global__ void k_facemax(const float* __restrict__ x, const int* __restrict__ v2f,
                          float* __restrict__ xf, int total) {
  int i = blockIdx.x * blockDim.x + threadIdx.x;
  if (i >= total) return;
  int p = i >> 9, c = i & 511;
  int vtx = v2f[p * 2 + 0];
  int fid = v2f[p * 2 + 1];
  atomicMaxF32(&xf[(size_t)fid * 512 + c], x[(size_t)vtx * 512 + c]);
}

// lin2 (512->8) + log_softmax, one thread per face
__global__ void k_lin2_ls(const float* __restrict__ in, const float* __restrict__ W,
                          const float* __restrict__ b, float* __restrict__ out) {
  int f = blockIdx.x * blockDim.x + threadIdx.x;
  if (f >= NF) return;
  float y[8];
  #pragma unroll
  for (int c = 0; c < 8; ++c) y[c] = b[c];
  const float* rowp = in + (size_t)f * 512;
  for (int k = 0; k < 512; ++k) {
    float a = rowp[k];
    #pragma unroll
    for (int c = 0; c < 8; ++c) y[c] += a * W[k * 8 + c];
  }
  float mx = y[0];
  #pragma unroll
  for (int c = 1; c < 8; ++c) mx = fmaxf(mx, y[c]);
  float s = 0.0f;
  #pragma unroll
  for (int c = 0; c < 8; ++c) s += __expf(y[c] - mx);
  float lse = mx + __logf(s);
  #pragma unroll
  for (int c = 0; c < 8; ++c) out[f * 8 + c] = y[c] - lse;
}

// ------------------------- gather-fused WMMA GEMM --------------------------
// OUT[M,Nc] = act( [gather(A1,idx1) | gather(A2,idx2)] @ W + bias + rank3 )
// W is bf16, pre-transposed [Nc][K] (K-contiguous). Block: 128 threads
// (4 waves), tile 16 rows x 64 cols, K-chunk 64 (2 WMMA per stage).
// B tile: async global->LDS b128 (ASYNCcnt). A tile: b128 load + bf16 pack.
// K1, K2 multiples of 64 (K2 may be 0); M % 16 == 0; Nc % 64 == 0.
__global__ __launch_bounds__(128)
void k_wmma_gemm(const float* __restrict__ A1, const int* __restrict__ idx1, int K1,
                 const float* __restrict__ A2, const int* __restrict__ idx2, int K2,
                 const unsigned short* __restrict__ WbT,
                 const float* __restrict__ bias,
                 const float* __restrict__ ev0, const float* __restrict__ ev1,
                 const float* __restrict__ ev2, const float* __restrict__ Wex, int ldex,
                 float* __restrict__ OUT, int Nc, int act) {
  __shared__ unsigned short As[16][64];      // 2 KB, row stride 128 B
  __shared__ unsigned short Bt[64][72];      // 9 KB, row stride 144 B (16B-aligned)
  const int tid  = threadIdx.x;
  const int lane = tid & 31;
  const int wave = tid >> 5;
  const int mIdx = lane & 15;                // A row / output column within tile
  const int g    = lane >> 4;                // lane-half select
  const int rowBase = blockIdx.x * 16;
  const int colBase = blockIdx.y * 64;
  const int K = K1 + K2;

  // Per-thread A-tile job: rows rA and rA+8, float4 chunk q. Gather indices
  // are loop-invariant -> load them once.
  const int rA = tid >> 4;                   // 0..7
  const int rB = rA + 8;
  const int q  = tid & 15;                   // float4 chunk within 64-wide K slab
  const int gA = rowBase + rA, gB = rowBase + rB;
  const int sA1 = idx1 ? idx1[gA] : gA;
  const int sB1 = idx1 ? idx1[gB] : gB;
  const int sA2 = idx2 ? idx2[gA] : gA;
  const int sB2 = idx2 ? idx2[gB] : gB;

  v8f acc = {};
  for (int k0 = 0; k0 < K; k0 += 64) {
    // A1/A2 selection is uniform per K-chunk (K1 % 64 == 0).
    const bool   in1 = (k0 < K1);
    const float* Ab  = in1 ? A1 : A2;
    const int    kb  = (in1 ? k0 : k0 - K1) + q * 4;
    const int    Ks  = in1 ? K1 : K2;
    const int    sA  = in1 ? sA1 : sA2;
    const int    sB  = in1 ? sB1 : sB2;

    // B tile: 64 cols x 64 K bf16, async global->LDS b128 copies (no VGPR
    // data path; tracked by ASYNCcnt). Low 32 bits of a flat LDS pointer are
    // the LDS byte address (ISA 10.2 aperture mapping).
    #pragma unroll
    for (int it = 0; it < 4; ++it) {
      int c = tid + it * 128;
      int colc = c >> 3, qq = c & 7;
      const unsigned short* gsrc = &WbT[(size_t)(colBase + colc) * K + k0 + qq * 8];
      unsigned ldsa = (unsigned)(unsigned long long)&Bt[colc][qq * 8];
      asm volatile("global_load_async_to_lds_b128 %0, %1, off"
                   :: "v"(ldsa), "v"(gsrc) : "memory");
    }

    // A tile: two b128 global loads -> bf16 pack -> two b64 LDS stores.
    const float4 va = *(const float4*)&Ab[(size_t)sA * Ks + kb];
    const float4 vb = *(const float4*)&Ab[(size_t)sB * Ks + kb];
    __builtin_prefetch(&Ab[(size_t)sA * Ks + kb + 64], 0, 0);  // next K-chunk
    *(uint2*)&As[rA][q * 4] = pack4_bf16(va);
    *(uint2*)&As[rB][q * 4] = pack4_bf16(vb);

    asm volatile("s_wait_asynccnt 0x0" ::: "memory");  // drain B-tile copies
    __syncthreads();

    // Two 16x16x32 WMMA steps; fragments via aligned b128 LDS loads.
    #pragma unroll
    for (int s = 0; s < 2; ++s) {
      v8bf alo = *(const v8bf*)&As[mIdx][s * 32 + 8 * g];
      v8bf ahi = *(const v8bf*)&As[mIdx][s * 32 + 16 + 8 * g];
      v8bf blo = *(const v8bf*)&Bt[wave * 16 + mIdx][s * 32 + 8 * g];
      v8bf bhi = *(const v8bf*)&Bt[wave * 16 + mIdx][s * 32 + 16 + 8 * g];
      v16bf a = __builtin_shufflevector(alo, ahi, 0, 1, 2, 3, 4, 5, 6, 7,
                                        8, 9, 10, 11, 12, 13, 14, 15);
      v16bf b = __builtin_shufflevector(blo, bhi, 0, 1, 2, 3, 4, 5, 6, 7,
                                        8, 9, 10, 11, 12, 13, 14, 15);
      acc = __builtin_amdgcn_wmma_f32_16x16x32_bf16(false, a, false, b,
                                                    (short)0, acc, false, false);
    }
    __syncthreads();
  }

  // Epilogue: bias + rank-3 edge tail + activation. D layout: M = r + 8*g.
  const int n = colBase + wave * 16 + mIdx;
  #pragma unroll
  for (int r = 0; r < 8; ++r) {
    int row = rowBase + 8 * g + r;
    float v = acc[r];
    if (bias) v += bias[n];
    if (Wex)  v += ev0[row] * Wex[n] + ev1[row] * Wex[ldex + n] + ev2[row] * Wex[2 * ldex + n];
    if (act == 1)      v = silu(v);
    else if (act == 2) v = fmaxf(v, 0.0f);
    OUT[(size_t)row * Nc + n] = v;
  }
}

// ---------------------------------------------------------------------------
extern "C" void kernel_launch(void* const* d_in, const int* in_sizes, int n_in,
                              void* d_out, int out_size, void* d_ws, size_t ws_size,
                              hipStream_t stream) {
  (void)in_sizes; (void)n_in; (void)out_size; (void)ws_size;
  const int N = NN, E = NE, F = NF;

  const float* pos  = (const float*)d_in[0];
  const float* area = (const float*)d_in[1];
  const float* hks  = (const float*)d_in[2];
  const float* wgt  = (const float*)d_in[3];
  const float* dia  = (const float*)d_in[4];
  const int*   eidx = (const int*)d_in[5];
  // d_in[6] = face (unused; vertex2face covers it)
  const int*   v2f  = (const int*)d_in[7];
  const int* row = eidx;        // edge_index[0]
  const int* col = eidx + E;    // edge_index[1]

  // params: JAX pytree (sorted-key) flattening order, starting at index 8:
  // conv1..conv3 (each: c1.W,c1.b,c2.W,e1.W,e1.b,e2.W,e2.b,n1.W,n1.b,n2.W,n2.b),
  // feat.W,feat.b, lin1.W,lin1.b, lin2.W,lin2.b
  struct Conv {
    const float *c1W, *c1b, *c2W, *e1W, *e1b, *e2W, *e2b, *n1W, *n1b, *n2W, *n2b;
    int d, hid, dout;
  } cv[3];
  const int dims[3][3] = {{64, 64, 128}, {128, 128, 256}, {256, 256, 512}};
  int p = 8;
  for (int l = 0; l < 3; ++l) {
    cv[l].c1W = (const float*)d_in[p++]; cv[l].c1b = (const float*)d_in[p++];
    cv[l].c2W = (const float*)d_in[p++];
    cv[l].e1W = (const float*)d_in[p++]; cv[l].e1b = (const float*)d_in[p++];
    cv[l].e2W = (const float*)d_in[p++]; cv[l].e2b = (const float*)d_in[p++];
    cv[l].n1W = (const float*)d_in[p++]; cv[l].n1b = (const float*)d_in[p++];
    cv[l].n2W = (const float*)d_in[p++]; cv[l].n2b = (const float*)d_in[p++];
    cv[l].d = dims[l][0]; cv[l].hid = dims[l][1]; cv[l].dout = dims[l][2];
  }
  const float* featW = (const float*)d_in[p++];
  const float* featb = (const float*)d_in[p++];
  const float* lin1W = (const float*)d_in[p++];
  const float* lin1b = (const float*)d_in[p++];
  const float* lin2W = (const float*)d_in[p++];
  const float* lin2b = (const float*)d_in[p++];

  // ---- workspace layout (floats, every block 16B aligned) ----
  float* wsf = (float*)d_ws;
  size_t o = 0;
  auto alloc = [&](size_t n) { float* r = wsf + o; o += n; return r; };
  float* x0       = alloc((size_t)N * 64);
  float* h1       = alloc((size_t)N * 128);
  float* h2       = alloc((size_t)N * 256);
  float* h3       = alloc((size_t)N * 512);
  float* coordA   = alloc((size_t)N * 3);
  float* coordB   = alloc((size_t)N * 3);
  float* coordAcc = alloc((size_t)N * 3);
  int*   cnt      = (int*)alloc((size_t)N);
  float* radial   = alloc((size_t)E);
  float* esc      = alloc((size_t)E);
  float* agg      = alloc((size_t)N * 256);
  float* nh       = alloc((size_t)N * 256);
  float* bufA     = alloc((size_t)E * 256);   // reused as xf [F,512] later
  float* bufB     = alloc((size_t)E * 256);   // reused as lin1 out later
  unsigned short* bfArena = (unsigned short*)alloc(600000); // 1.2M bf16 slots

  auto g1 = [](int n) { return dim3((unsigned)((n + 255) / 256)); };

  // ---- weight conversion fp32 -> bf16, transposed to [Nc][K] ----
  size_t bo = 0;
  auto cvtT = [&](const float* src, int K, int Nc) {
    unsigned short* dst = bfArena + bo; bo += (size_t)K * Nc;
    hipLaunchKernelGGL(k_f32_to_bf16_T, g1(K * Nc), dim3(256), 0, stream, src, dst, K, Nc);
    return dst;
  };
  unsigned short *e1b16[3], *e2b16[3], *c1b16[3], *n1b16[3], *n2b16[3];
  for (int l = 0; l < 3; ++l) {
    int d = cv[l].d, hid = cv[l].hid, dout = cv[l].dout;
    e1b16[l] = cvtT(cv[l].e1W, 2 * d, hid);   // +3 tail rows stay fp32 (epilogue)
    e2b16[l] = cvtT(cv[l].e2W, hid, hid);
    c1b16[l] = cvtT(cv[l].c1W, hid, hid);
    n1b16[l] = cvtT(cv[l].n1W, d + hid, hid);
    n2b16[l] = cvtT(cv[l].n2W, hid, dout);
  }
  unsigned short* lin1b16 = cvtT(lin1W, 512, 512);

  auto gemm = [&](const float* A1, const int* i1, int K1,
                  const float* A2, const int* i2, int K2,
                  const unsigned short* WbT, const float* bias,
                  const float* ev0, const float* ev1, const float* ev2,
                  const float* Wex, float* OUTp, int M, int Nc, int act) {
    dim3 grid((unsigned)(M / 16), (unsigned)(Nc / 64));
    hipLaunchKernelGGL(k_wmma_gemm, grid, dim3(128), 0, stream,
                       A1, i1, K1, A2, i2, K2, WbT, bias,
                       ev0, ev1, ev2, Wex, Nc, OUTp, Nc, act);
  };

  // ---- node feature lift + static graph prep ----
  hipLaunchKernelGGL(k_feat, g1(N * 64), dim3(256), 0, stream, area, hks, featW, featb, x0);
  hipLaunchKernelGGL(k_copy_f32, g1(N * 3), dim3(256), 0, stream, pos, coordA, N * 3);
  hipLaunchKernelGGL(k_zero_i32, g1(N), dim3(256), 0, stream, cnt, N);
  hipLaunchKernelGGL(k_cnt, g1(E), dim3(256), 0, stream, row, cnt);

  const float* hcur = x0;
  float* hnext[3] = {h1, h2, h3};
  float* cc = coordA; float* cn = coordB;

  for (int l = 0; l < 3; ++l) {
    int d = cv[l].d, hid = cv[l].hid, dout = cv[l].dout;

    hipLaunchKernelGGL(k_radial, g1(E), dim3(256), 0, stream, cc, row, col, radial);

    // m1 = silu(e1([h[row] | h[col] | radial | weight | di]))  -- gather-fused
    gemm(hcur, row, d, hcur, col, d, e1b16[l], cv[l].e1b,
         radial, wgt, dia, cv[l].e1W + (size_t)(2 * d) * hid,
         bufA, E, hid, 1);
    // m = silu(e2(m1))
    gemm(bufA, nullptr, hid, nullptr, nullptr, 0, e2b16[l], cv[l].e2b,
         nullptr, nullptr, nullptr, nullptr, bufB, E, hid, 1);

    if (l < 2) {  // conv3's coordinate branch never reaches the output
      gemm(bufB, nullptr, hid, nullptr, nullptr, 0, c1b16[l], cv[l].c1b,
           nullptr, nullptr, nullptr, nullptr, bufA, E, hid, 1);
      hipLaunchKernelGGL(k_esc, g1(E), dim3(256), 0, stream, bufA, cv[l].c2W, hid, esc);
      hipLaunchKernelGGL(k_zero_f32, g1(N * 3), dim3(256), 0, stream, coordAcc, N * 3);
      hipLaunchKernelGGL(k_coord_scatter, g1(E), dim3(256), 0, stream, cc, row, col, esc, coordAcc);
      hipLaunchKernelGGL(k_coord_update, g1(N * 3), dim3(256), 0, stream, cc, coordAcc, cnt, cn);
    }

    // agg = segment_sum(m, row)
    hipLaunchKernelGGL(k_zero_f32, g1(N * hid), dim3(256), 0, stream, agg, N * hid);
    hipLaunchKernelGGL(k_agg_scatter, g1(E * hid), dim3(256), 0, stream, bufB, row, hid, agg, E * hid);

    // h_out = n2(silu(n1([h | agg])))
    gemm(hcur, nullptr, d, agg, nullptr, hid, n1b16[l], cv[l].n1b,
         nullptr, nullptr, nullptr, nullptr, nh, N, hid, 1);
    gemm(nh, nullptr, hid, nullptr, nullptr, 0, n2b16[l], cv[l].n2b,
         nullptr, nullptr, nullptr, nullptr, hnext[l], N, dout, 0);

    hcur = hnext[l];
    if (l < 2) { float* t = cc; cc = cn; cn = t; }
  }

  // ---- face max pooling (scatter-max) ----
  float* xf = bufA;  // [F,512]
  hipLaunchKernelGGL(k_fill_bits, g1(F * 512), dim3(256), 0, stream,
                     (unsigned int*)xf, 0xFF7FFFFFu /* -FLT_MAX */, F * 512);
  hipLaunchKernelGGL(k_facemax, g1(3 * F * 512), dim3(256), 0, stream, h3, v2f, xf, 3 * F * 512);

  // ---- head: lin1 (WMMA, relu) + lin2 + log_softmax ----
  float* lo = bufB;  // [F,512]
  gemm(xf, nullptr, 512, nullptr, nullptr, 0, lin1b16, lin1b,
       nullptr, nullptr, nullptr, nullptr, lo, F, 512, 2);
  hipLaunchKernelGGL(k_lin2_ls, g1(F), dim3(256), 0, stream, lo, lin2W, lin2b, (float*)d_out);
}
